// Attention_12962211299892
// MI455X (gfx1250) — matrix operand order repaired and checked by reading, not compiled, not run on hardware
//
#include <hip/hip_runtime.h>

typedef _Float16 h16;
typedef __attribute__((ext_vector_type(8)))  _Float16 v8h;
typedef __attribute__((ext_vector_type(16))) _Float16 v16h;
typedef __attribute__((ext_vector_type(8)))  float    v8f;
typedef __attribute__((ext_vector_type(4)))  unsigned int v4u;
typedef __attribute__((ext_vector_type(8)))  int      v8i_;
typedef __attribute__((ext_vector_type(4)))  int      v4i_;

#define B_   2
#define S_   2048
#define HID_ 1024
#define NH_  16
#define HD_  64
#define MTOT (B_ * S_)          // 4096 tokens

#define AS3 __attribute__((address_space(3)))

// ---------------------------------------------------------------------------
// WMMA fragment helpers (layouts per CDNA5 ISA 7.12.2, wave32)
// ---------------------------------------------------------------------------
__device__ __forceinline__ v16h frag_cat(v8h a, v8h b) {
  return __builtin_shufflevector(a, b, 0,1,2,3,4,5,6,7,8,9,10,11,12,13,14,15);
}

// A-matrix 16x32 f16 fragment from row-major A[16][lda] (global or LDS).
__device__ __forceinline__ v16h load_frag_A(const h16* A, int lda, int k0, int lane) {
  int m = lane & 15, hf = lane >> 4;
  const h16* p = A + (size_t)m * lda + k0 + 8 * hf;
  v8h a0 = *(const v8h*)p;
  v8h a1 = *(const v8h*)(p + 16);
  return frag_cat(a0, a1);
}

// B-matrix 32x16 f16 fragment from Bt = B^T row-major [16 n][ldb>=k0+32].
__device__ __forceinline__ v16h load_frag_Bt(const h16* Bt, int ldb, int k0, int lane) {
  int n = lane & 15, hf = lane >> 4;
  const h16* p = Bt + (size_t)n * ldb + k0 + 8 * hf;
  v8h b0 = *(const v8h*)p;
  v8h b1 = *(const v8h*)(p + 16);
  return frag_cat(b0, b1);
}

__device__ __forceinline__ v8f wmma16(v16h a, v16h b, v8f c) {
  return __builtin_amdgcn_wmma_f32_16x16x32_f16(false, a, false, b, (short)0, c, false, false);
}

// ---------------------------------------------------------------------------
// Tensor Data Mover: 2D tile (f16) global -> LDS.  D# per ISA 08 §8.
// rowlen  = tile row length in elements (= tensor_dim0 = tile_dim0)
// nrows   = number of rows (= tensor_dim1 = tile_dim1)
// rstride = row stride in elements (tensor_dim0_stride)
// ---------------------------------------------------------------------------
__device__ __forceinline__ unsigned lds_byte_off(const void* p) {
  return (unsigned)(unsigned long long)(AS3 const char*)p;
}

__device__ __forceinline__ void tdm_load_2d(unsigned ldsoff, const h16* gsrc,
                                            unsigned rowlen, unsigned nrows,
                                            unsigned rstride) {
  unsigned long long ga = (unsigned long long)(const void*)gsrc;
  v4u g0;
  g0.x = 1u;                                                  // count=1
  g0.y = ldsoff;                                              // lds_addr
  g0.z = (unsigned)ga;                                        // global_addr[31:0]
  g0.w = (unsigned)((ga >> 32) & 0x01ffffffull) | (2u << 30); // ga[56:32] | type=2
  v8i_ g1;
  g1[0] = 0x00010000;                                         // data_size=2B, mask=0
  g1[1] = (int)((rowlen & 0xffffu) << 16);                    // tensor_dim0 lo
  g1[2] = (int)(((rowlen >> 16) & 0xffffu) | ((nrows & 0xffffu) << 16));
  g1[3] = (int)(((nrows >> 16) & 0xffffu) | ((rowlen & 0xffffu) << 16)); // tile_dim0
  g1[4] = (int)(nrows & 0xffffu);                             // tile_dim1, tile_dim2=0
  g1[5] = (int)rstride;                                       // dim0 stride lo32
  g1[6] = 0;
  g1[7] = 0;
  v4i_ z4 = {0, 0, 0, 0};
#if __clang_major__ >= 23
  v8i_ z8 = {0, 0, 0, 0, 0, 0, 0, 0};
  __builtin_amdgcn_tensor_load_to_lds(g0, g1, z4, z4, z8, 0);
#else
  __builtin_amdgcn_tensor_load_to_lds(g0, g1, z4, z4, 0);
#endif
}

// ---------------------------------------------------------------------------
// f32 -> f16 conversion
// ---------------------------------------------------------------------------
__global__ __launch_bounds__(256) void cvt_f32_f16(const float* __restrict__ src,
                                                   h16* __restrict__ dst, int n) {
  int i = blockIdx.x * 256 + threadIdx.x;
  if (i < n) dst[i] = (h16)src[i];
}

// ---------------------------------------------------------------------------
// GEMM: C[M][N] (f32) = A[M][K] (f16) @ Bt[N][K]^T (f16) + bias[N]
// one wave -> 16x64 strip; register double-buffered K-loop so fragment loads
// for step k+32 overlap the 4 WMMAs of step k.
// ---------------------------------------------------------------------------
__global__ __launch_bounds__(256) void gemm_xwT(const h16* __restrict__ A,
                                                const h16* __restrict__ Bt,
                                                const float* __restrict__ bias,
                                                float* __restrict__ C,
                                                int M, int N, int K) {
  int lane = threadIdx.x & 31;
  int wave = (blockIdx.x * blockDim.x + threadIdx.x) >> 5;
  int ntiles = N >> 6;
  int mt = wave / ntiles;
  int nt = wave - mt * ntiles;
  if (mt * 16 >= M) return;

  const h16* Arow = A + (size_t)mt * 16 * K;
  const h16* B0   = Bt + (size_t)nt * 64 * K;
  v8f acc[4] = {};

  v16h a = load_frag_A(Arow, K, 0, lane);
  v16h b[4];
#pragma unroll
  for (int j = 0; j < 4; ++j) b[j] = load_frag_Bt(B0 + (size_t)j * 16 * K, K, 0, lane);

  for (int k0 = 0; k0 < K; k0 += 32) {
    v16h an = a;
    v16h bn0 = b[0], bn1 = b[1], bn2 = b[2], bn3 = b[3];
    int k1 = k0 + 32;
    if (k1 < K) {               // prefetch next step's fragments
      an  = load_frag_A(Arow, K, k1, lane);
      bn0 = load_frag_Bt(B0 + (size_t)0 * 16 * K, K, k1, lane);
      bn1 = load_frag_Bt(B0 + (size_t)1 * 16 * K, K, k1, lane);
      bn2 = load_frag_Bt(B0 + (size_t)2 * 16 * K, K, k1, lane);
      bn3 = load_frag_Bt(B0 + (size_t)3 * 16 * K, K, k1, lane);
    }
    acc[0] = wmma16(a, b[0], acc[0]);
    acc[1] = wmma16(a, b[1], acc[1]);
    acc[2] = wmma16(a, b[2], acc[2]);
    acc[3] = wmma16(a, b[3], acc[3]);
    a = an; b[0] = bn0; b[1] = bn1; b[2] = bn2; b[3] = bn3;
  }

  int n = lane & 15, hf = lane >> 4;
#pragma unroll
  for (int j = 0; j < 4; ++j) {
    int col = nt * 64 + j * 16 + n;
    float bv = bias ? bias[col] : 0.0f;
#pragma unroll
    for (int r = 0; r < 8; ++r) {
      int row = mt * 16 + r + 8 * hf;
      C[(size_t)row * N + col] = acc[j][r] + bv;
    }
  }
}

// ---------------------------------------------------------------------------
// RoPE on f32 Q/K raw ([tok][NH*HD]) -> f16 packed [B][NH][S][HD]
// ---------------------------------------------------------------------------
__global__ __launch_bounds__(256) void rope_kernel(const float* __restrict__ Qr,
                                                   const float* __restrict__ Kr,
                                                   h16* __restrict__ Qh,
                                                   h16* __restrict__ Kh) {
  int idx = blockIdx.x * 256 + threadIdx.x;     // B*NH*S*32 threads
  int j  = idx & 31;
  int s  = (idx >> 5) & (S_ - 1);
  int t  = idx >> 16;
  int hh = t & (NH_ - 1);
  int b  = t >> 4;

  float ang = (float)s * exp2f(-0.5f * (float)j);   // 65536^(-2j/64) = 2^(-j/2)
  float sn, c;
  sincosf(ang, &sn, &c);

  size_t ib = ((size_t)(b * S_ + s)) * (NH_ * HD_) + hh * HD_;
  size_t ob = (((size_t)(b * NH_ + hh)) * S_ + s) * HD_;

  float q1 = Qr[ib + j], q2 = Qr[ib + j + 32];
  Qh[ob + j]      = (h16)(q1 * c - q2 * sn);
  Qh[ob + j + 32] = (h16)(q2 * c + q1 * sn);
  float k1 = Kr[ib + j], k2 = Kr[ib + j + 32];
  Kh[ob + j]      = (h16)(k1 * c - k2 * sn);
  Kh[ob + j + 32] = (h16)(k2 * c + k1 * sn);
}

// V raw f32 [tok][NH*HD] -> f16 transposed [B][NH][HD][S]
__global__ __launch_bounds__(256) void vtrans_kernel(const float* __restrict__ Vr,
                                                     h16* __restrict__ Vt) {
  int idx = blockIdx.x * 256 + threadIdx.x;     // B*NH*S*HD threads
  int d  = idx & 63;
  int s  = (idx >> 6) & (S_ - 1);
  int t  = idx >> 17;
  int hh = t & (NH_ - 1);
  int b  = t >> 4;
  Vt[(((size_t)(b * NH_ + hh)) * HD_ + d) * S_ + s] =
      (h16)Vr[((size_t)(b * S_ + s)) * (NH_ * HD_) + hh * HD_ + d];
}

// ---------------------------------------------------------------------------
// Flash attention: one wave per (b, h, 16-query tile).
// K-tile (32x64) and V-tile (64x32) staged in LDS by the Tensor Data Mover,
// double buffered: TDM for block i+1 overlaps WMMA compute on block i.
// P goes through LDS for the C-layout -> A-layout transpose.
// ---------------------------------------------------------------------------
__global__ __launch_bounds__(256) void attn_fa(const h16* __restrict__ Qh,
                                               const h16* __restrict__ Kh,
                                               const h16* __restrict__ Vt,
                                               h16* __restrict__ Oh) {
  __shared__ __align__(16) h16 KT[8][2][32 * 64];   // [wave][buf][key][d]
  __shared__ __align__(16) h16 VT[8][2][64 * 32];   // [wave][buf][d][key]
  __shared__ __align__(16) h16 Pt[8][16 * 32];      // [wave][m][k]

  int wid  = threadIdx.x >> 5;
  int lane = threadIdx.x & 31;
  int gw   = blockIdx.x * 8 + wid;

  int qt = gw & 127;                            // S/16 = 128
  int t  = gw >> 7;
  int hh = t & (NH_ - 1);
  int b  = t >> 4;

  const h16* Qtile = Qh + (((size_t)(b * NH_ + hh)) * S_ + qt * 16) * HD_;
  const h16* Kbase = Kh + ((size_t)(b * NH_ + hh)) * S_ * HD_;
  const h16* Vtb   = Vt + ((size_t)(b * NH_ + hh)) * HD_ * S_;   // [d][s], ld=S
  h16* P = &Pt[wid][0];

  const h16* KTb[2] = { &KT[wid][0][0], &KT[wid][1][0] };
  const h16* VTb[2] = { &VT[wid][0][0], &VT[wid][1][0] };
  unsigned koff[2] = { lds_byte_off(KTb[0]), lds_byte_off(KTb[1]) };
  unsigned voff[2] = { lds_byte_off(VTb[0]), lds_byte_off(VTb[1]) };

  v16h qa0 = load_frag_A(Qtile, HD_, 0,  lane);
  v16h qa1 = load_frag_A(Qtile, HD_, 32, lane);

  v8f  accO[4] = {};
  float mrow[8], lrow[8];
#pragma unroll
  for (int r = 0; r < 8; ++r) { mrow[r] = -1e30f; lrow[r] = 0.0f; }

  const float scale = 0.125f;                   // 1/sqrt(64)
  int n = lane & 15, hf = lane >> 4;

  // preload key-block 0 via TDM
  tdm_load_2d(koff[0], Kbase, 64, 32, 64);      // 32 key rows x 64 dims
  tdm_load_2d(voff[0], Vtb,   32, 64, S_);      // 64 dim rows x 32 keys
  __builtin_amdgcn_s_wait_tensorcnt((short)0);

  for (int kb = 0; kb < S_; kb += 32) {
    int cur = (kb >> 5) & 1;
    if (kb + 32 < S_) {                         // overlap next block's DMA
      tdm_load_2d(koff[cur ^ 1], Kbase + (size_t)(kb + 32) * HD_, 64, 32, 64);
      tdm_load_2d(voff[cur ^ 1], Vtb + (kb + 32), 32, 64, S_);
    }
    const h16* Kt = KTb[cur];
    const h16* Vl = VTb[cur];

    // ---- scores: two 16x16 tiles (keys 0..15, 16..31 of block) from LDS
    v8f s0 = {}, s1 = {};
    s0 = wmma16(qa0, load_frag_Bt(Kt,            64, 0,  lane), s0);
    s0 = wmma16(qa1, load_frag_Bt(Kt,            64, 32, lane), s0);
    s1 = wmma16(qa0, load_frag_Bt(Kt + 16 * 64,  64, 0,  lane), s1);
    s1 = wmma16(qa1, load_frag_Bt(Kt + 16 * 64,  64, 32, lane), s1);

    // ---- online softmax (row M = r+8*half spans the 16 lanes of a half-group)
#pragma unroll
    for (int r = 0; r < 8; ++r) {
      float v0 = s0[r] * scale, v1 = s1[r] * scale;
      float mx = fmaxf(v0, v1);
#pragma unroll
      for (int off = 8; off > 0; off >>= 1)
        mx = fmaxf(mx, __shfl_xor(mx, off, 16));
      float mnew = fmaxf(mrow[r], mx);
      float p0 = expf(v0 - mnew), p1 = expf(v1 - mnew);
      float psum = p0 + p1;
#pragma unroll
      for (int off = 8; off > 0; off >>= 1)
        psum += __shfl_xor(psum, off, 16);
      float corr = expf(mrow[r] - mnew);
      lrow[r] = lrow[r] * corr + psum;
      mrow[r] = mnew;
#pragma unroll
      for (int j = 0; j < 4; ++j) accO[j][r] *= corr;
      P[(r + 8 * hf) * 32 + n]      = (h16)p0;
      P[(r + 8 * hf) * 32 + n + 16] = (h16)p1;
    }
    __builtin_amdgcn_wave_barrier();            // ds stores before ds loads

    // ---- P @ V for this block (V tile from LDS)
    v16h pa = load_frag_A(P, 32, 0, lane);
#pragma unroll
    for (int j = 0; j < 4; ++j) {
      v16h bv = load_frag_Bt(Vl + (size_t)j * 16 * 32, 32, 0, lane);
      accO[j] = wmma16(pa, bv, accO[j]);
    }
    __builtin_amdgcn_wave_barrier();

    // next block's TDM transfers must have landed before we read them
    __builtin_amdgcn_s_wait_tensorcnt((short)0);
  }

  // ---- normalize and store f16 attn output [tok][NH*HD]
#pragma unroll
  for (int j = 0; j < 4; ++j) {
#pragma unroll
    for (int r = 0; r < 8; ++r) {
      int row = qt * 16 + r + 8 * hf;
      float o = accO[j][r] * (1.0f / lrow[r]);
      Oh[((size_t)(b * S_ + row)) * (NH_ * HD_) + hh * HD_ + j * 16 + n] = (h16)o;
    }
  }
}

// ---------------------------------------------------------------------------
// Host-side orchestration
// ---------------------------------------------------------------------------
extern "C" void kernel_launch(void* const* d_in, const int* in_sizes, int n_in,
                              void* d_out, int out_size, void* d_ws, size_t ws_size,
                              hipStream_t stream) {
  (void)in_sizes; (void)n_in; (void)out_size; (void)ws_size;
  const float* X  = (const float*)d_in[0];
  const float* Wq = (const float*)d_in[1];
  const float* bq = (const float*)d_in[2];
  const float* Wk = (const float*)d_in[3];
  const float* bk = (const float*)d_in[4];
  const float* Wv = (const float*)d_in[5];
  const float* bv = (const float*)d_in[6];
  const float* Wo = (const float*)d_in[7];
  const float* bo = (const float*)d_in[8];
  float* out = (float*)d_out;

  const size_t nX   = (size_t)MTOT * HID_;
  const size_t nW   = (size_t)HID_ * HID_;
  const size_t nQKV = (size_t)B_ * NH_ * S_ * HD_;

  h16* Xh  = (h16*)d_ws;
  h16* Wqh = Xh  + nX;
  h16* Wkh = Wqh + nW;
  h16* Wvh = Wkh + nW;
  h16* Woh = Wvh + nW;
  h16* Qh  = Woh + nW;
  h16* Kh  = Qh  + nQKV;
  h16* Vth = Kh  + nQKV;
  h16* Oh  = Vth + nQKV;
  float* Qraw = (float*)(Oh + nX);
  float* Kraw = Qraw + nX;
  float* Vraw = Kraw + nX;

  cvt_f32_f16<<<(int)(nX / 256), 256, 0, stream>>>(X,  Xh,  (int)nX);
  cvt_f32_f16<<<(int)(nW / 256), 256, 0, stream>>>(Wq, Wqh, (int)nW);
  cvt_f32_f16<<<(int)(nW / 256), 256, 0, stream>>>(Wk, Wkh, (int)nW);
  cvt_f32_f16<<<(int)(nW / 256), 256, 0, stream>>>(Wv, Wvh, (int)nW);
  cvt_f32_f16<<<(int)(nW / 256), 256, 0, stream>>>(Wo, Woh, (int)nW);

  gemm_xwT<<<512, 256, 0, stream>>>(Xh, Wqh, bq, Qraw, MTOT, HID_, HID_);
  gemm_xwT<<<512, 256, 0, stream>>>(Xh, Wkh, bk, Kraw, MTOT, HID_, HID_);
  gemm_xwT<<<512, 256, 0, stream>>>(Xh, Wvh, bv, Vraw, MTOT, HID_, HID_);

  rope_kernel<<<(B_ * NH_ * S_ * 32) / 256, 256, 0, stream>>>(Qraw, Kraw, Qh, Kh);
  vtrans_kernel<<<(B_ * NH_ * S_ * HD_) / 256, 256, 0, stream>>>(Vraw, Vth);

  attn_fa<<<512, 256, 0, stream>>>(Qh, Kh, Vth, Oh);

  gemm_xwT<<<512, 256, 0, stream>>>(Oh, Woh, bo, out, MTOT, HID_, HID_);
}
